// GraphEncoder_78202764525698
// MI455X (gfx1250) — compile-verified, matrix operand order checked by reading, not run on hardware
//
#include <hip/hip_runtime.h>
#include <hip/hip_bf16.h>

#define NN 20000
#define EE 320000
#define E2 340000   // EE + NN self loops
#define IND 384
#define DD 256
#define HH 4
#define CC 64
#define LL 3
#define NT 4        // N-tiles per wave in GEMM

typedef __attribute__((ext_vector_type(16))) __bf16 v16bf;
typedef __attribute__((ext_vector_type(8)))  float  v8f;

union Frag { float4 q[2]; v16bf v; };

__device__ __forceinline__ unsigned short f2bf(float f) {
    unsigned u = __float_as_uint(f);
    u += 0x7FFFu + ((u >> 16) & 1u);          // round-to-nearest-even
    return (unsigned short)(u >> 16);
}

__device__ __forceinline__ float lrelu(float v) { return v > 0.f ? v : 0.2f * v; }

__device__ __forceinline__ int atomAddI(int* p, int v) {
    return __hip_atomic_fetch_add(p, v, __ATOMIC_RELAXED, __HIP_MEMORY_SCOPE_AGENT);
}

// ---------------------------------------------------------------------------
__global__ void __launch_bounds__(256)
f32_to_bf16_k(const float* __restrict__ x, unsigned short* __restrict__ y, int n) {
    int i = blockIdx.x * blockDim.x + threadIdx.x;
    if (i < n) y[i] = f2bf(x[i]);
}

// W[K][Nc] f32 -> WT[Nc][K] bf16 (transpose so B fragments load contiguously)
__global__ void __launch_bounds__(256)
wT_bf16_k(const float* __restrict__ W, unsigned short* __restrict__ WT, int K, int Nc) {
    int i = blockIdx.x * blockDim.x + threadIdx.x;
    if (i >= K * Nc) return;
    int k = i / Nc, n = i % Nc;
    WT[(size_t)n * K + k] = f2bf(W[i]);
}

// ---------------------------------------------------------------------------
// C[M][256] = A[M][K](bf16) * BT[256][K](bf16) (+bias).  8 waves/block:
// waves 0-3 cover M-tile 2*bx, waves 4-7 cover 2*bx+1; each wave owns 4
// consecutive 16-col tiles, reusing its A fragment across 4 WMMAs per K step.
__global__ void __launch_bounds__(256)
gemm_bf16_wmma(const unsigned short* __restrict__ A,
               const unsigned short* __restrict__ BT,
               const float* __restrict__ bias,
               float* __restrict__ C,
               int Ncols, int K) {
    const int lane  = threadIdx.x & 31;
    const int khalf = lane >> 4;
    const int idx   = lane & 15;
    const int wave  = threadIdx.x >> 5;
    const int mt  = blockIdx.x * 2 + (wave >> 2);
    const int ntb = (wave & 3) * NT;

    const unsigned short* Arow  = A  + (size_t)(mt * 16 + idx) * K;
    const unsigned short* Brow0 = BT + (size_t)(ntb * 16 + idx) * K;

    v8f acc[NT];
#pragma unroll
    for (int t = 0; t < NT; ++t) acc[t] = (v8f){0.f,0.f,0.f,0.f,0.f,0.f,0.f,0.f};

    for (int k = 0; k < K; k += 32) {
        const int off = k + khalf * 8;
        Frag a;
        a.q[0] = *reinterpret_cast<const float4*>(Arow + off);
        a.q[1] = *reinterpret_cast<const float4*>(Arow + off + 16);
#pragma unroll
        for (int t = 0; t < NT; ++t) {
            Frag b;
            const unsigned short* Br = Brow0 + (size_t)(t * 16) * K;
            b.q[0] = *reinterpret_cast<const float4*>(Br + off);
            b.q[1] = *reinterpret_cast<const float4*>(Br + off + 16);
            acc[t] = __builtin_amdgcn_wmma_f32_16x16x32_bf16(
                         false, a.v, false, b.v, (short)0, acc[t], false, false);
        }
    }

    const int r0 = mt * 16 + khalf * 8;   // D layout: VGPR v -> row v + half*8
#pragma unroll
    for (int t = 0; t < NT; ++t) {
        const int col = (ntb + t) * 16 + idx;
        const float bv = bias ? bias[col] : 0.f;
#pragma unroll
        for (int v = 0; v < 8; ++v)
            C[(size_t)(r0 + v) * Ncols + col] = acc[t][v] + bv;
    }
}

// ---------------------------------------------------------------------------
// CSR-by-destination build (edge topology constant across layers)
__global__ void __launch_bounds__(256)
zero_int2_k(int* __restrict__ a, int* __restrict__ b2, int n) {
    int i = blockIdx.x * blockDim.x + threadIdx.x;
    if (i < n) { a[i] = 0; b2[i] = 0; }
}

__global__ void __launch_bounds__(256)
csr_count_k(const int* __restrict__ ei, int* __restrict__ deg) {
    int t = blockIdx.x * blockDim.x + threadIdx.x;
    if (t >= E2) return;
    int d = (t < EE) ? ei[EE + t] : t - EE;
    atomAddI(&deg[d], 1);
}

// single-block exclusive scan of deg[NN] -> rowoff[NN]
__global__ void __launch_bounds__(256)
csr_scan_k(const int* __restrict__ deg, int* __restrict__ rowoff) {
    __shared__ int part[256];
    const int t = threadIdx.x;
    const int CH = (NN + 255) / 256;
    const int base = t * CH;
    int sum = 0;
    for (int i = 0; i < CH; ++i) { int n = base + i; if (n < NN) sum += deg[n]; }
    part[t] = sum;
    __syncthreads();
    for (int o = 1; o < 256; o <<= 1) {
        int x = (t >= o) ? part[t - o] : 0;
        __syncthreads();
        part[t] += x;
        __syncthreads();
    }
    int run = part[t] - sum;   // exclusive prefix
    for (int i = 0; i < CH; ++i) {
        int n = base + i;
        if (n < NN) { rowoff[n] = run; run += deg[n]; }
    }
}

__global__ void __launch_bounds__(256)
csr_fill_k(const int* __restrict__ ei, const int* __restrict__ rowoff,
           int* __restrict__ cursor, int* __restrict__ elist) {
    int t = blockIdx.x * blockDim.x + threadIdx.x;
    if (t >= E2) return;
    int d = (t < EE) ? ei[EE + t] : t - EE;
    int pos = atomAddI(&cursor[d], 1);
    elist[rowoff[d] + pos] = t;
}

// ---------------------------------------------------------------------------
// Fused GATv2 edge phase + bias + LayerNorm + residual.  One wave32 per
// destination node; channels c = lane + 32*j (j=0..7); head of j is j>>1.
// No float atomics: max/exp/sum/aggregate all sequential per node.
__global__ void __launch_bounds__(256)
gat_node_k(const float* __restrict__ xl, const float* __restrict__ xr,
           const int* __restrict__ ei, const int* __restrict__ rowoff,
           const int* __restrict__ deg, const int* __restrict__ elist,
           const float* __restrict__ att_l, const float* __restrict__ bias_l,
           const float* __restrict__ g, const float* __restrict__ b,
           float* __restrict__ h, float* __restrict__ ebuf) {
    const int node = blockIdx.x * 8 + (threadIdx.x >> 5);
    const int lane = threadIdx.x & 31;
    if (node >= NN) return;

    float xrv[8], attv[8];
#pragma unroll
    for (int j = 0; j < 8; ++j) {
        int c = lane + 32 * j;
        xrv[j]  = xr[(size_t)node * DD + c];
        attv[j] = att_l[c];
    }
    const int start = rowoff[node];
    const int cnt   = deg[node];

    // Pass 1: per-edge logits for all 4 heads; running max in registers
    float mx[4] = {-3.4e38f, -3.4e38f, -3.4e38f, -3.4e38f};
    for (int k = 0; k < cnt; ++k) {
        const int eidx = elist[start + k];
        const int s = (eidx < EE) ? ei[eidx] : eidx - EE;
        const float* xs = xl + (size_t)s * DD;
        float p[4] = {0.f, 0.f, 0.f, 0.f};
#pragma unroll
        for (int j = 0; j < 8; ++j)
            p[j >> 1] += lrelu(xs[lane + 32 * j] + xrv[j]) * attv[j];
#pragma unroll
        for (int i = 0; i < 4; ++i) {
#pragma unroll
            for (int o = 16; o > 0; o >>= 1) p[i] += __shfl_xor(p[i], o, 32);
            mx[i] = fmaxf(mx[i], p[i]);
        }
        if (lane == 0)
            *reinterpret_cast<float4*>(ebuf + (size_t)(start + k) * 4) =
                make_float4(p[0], p[1], p[2], p[3]);
    }

    // Pass 2: softmax denominators
    float se[4] = {0.f, 0.f, 0.f, 0.f};
    for (int k = 0; k < cnt; ++k) {
        float4 lv = *reinterpret_cast<const float4*>(ebuf + (size_t)(start + k) * 4);
        se[0] += __expf(lv.x - mx[0]);
        se[1] += __expf(lv.y - mx[1]);
        se[2] += __expf(lv.z - mx[2]);
        se[3] += __expf(lv.w - mx[3]);
    }
    float rse[4];
#pragma unroll
    for (int i = 0; i < 4; ++i) rse[i] = 1.f / (se[i] + 1e-16f);

    // Pass 3: weighted aggregation (+ bias init)
    float acc[8];
#pragma unroll
    for (int j = 0; j < 8; ++j) acc[j] = bias_l[lane + 32 * j];
    for (int k = 0; k < cnt; ++k) {
        const int eidx = elist[start + k];
        const int s = (eidx < EE) ? ei[eidx] : eidx - EE;
        float4 lv = *reinterpret_cast<const float4*>(ebuf + (size_t)(start + k) * 4);
        float aa[4];
        aa[0] = __expf(lv.x - mx[0]) * rse[0];
        aa[1] = __expf(lv.y - mx[1]) * rse[1];
        aa[2] = __expf(lv.z - mx[2]) * rse[2];
        aa[3] = __expf(lv.w - mx[3]) * rse[3];
        const float* xs = xl + (size_t)s * DD;
#pragma unroll
        for (int j = 0; j < 8; ++j)
            acc[j] += xs[lane + 32 * j] * aa[j >> 1];
    }

    // Fused LayerNorm + residual
    float sum = 0.f;
#pragma unroll
    for (int j = 0; j < 8; ++j) sum += acc[j];
#pragma unroll
    for (int o = 16; o > 0; o >>= 1) sum += __shfl_xor(sum, o, 32);
    const float mu = sum * (1.f / DD);
    float vs = 0.f;
#pragma unroll
    for (int j = 0; j < 8; ++j) { float t = acc[j] - mu; vs += t * t; }
#pragma unroll
    for (int o = 16; o > 0; o >>= 1) vs += __shfl_xor(vs, o, 32);
    const float rstd = rsqrtf(vs * (1.f / DD) + 1e-5f);
    float* hrow = h + (size_t)node * DD;
#pragma unroll
    for (int j = 0; j < 8; ++j) {
        int c = lane + 32 * j;
        hrow[c] += (acc[j] - mu) * rstd * g[c] + b[c];
    }
}

// one wave per node: out = h / max(||h||, 1e-12)
__global__ void __launch_bounds__(256)
final_norm_k(const float* __restrict__ h, float* __restrict__ out) {
    int node = blockIdx.x * 8 + (threadIdx.x >> 5);
    int lane = threadIdx.x & 31;
    if (node >= NN) return;
    const float* row = h + (size_t)node * DD;
    float v[8];
    float s = 0.f;
#pragma unroll
    for (int j = 0; j < 8; ++j) { v[j] = row[lane + 32 * j]; s += v[j] * v[j]; }
#pragma unroll
    for (int o = 16; o > 0; o >>= 1) s += __shfl_xor(s, o, 32);
    float inv = 1.f / fmaxf(sqrtf(s), 1e-12f);
    float* orow = out + (size_t)node * DD;
#pragma unroll
    for (int j = 0; j < 8; ++j) orow[lane + 32 * j] = v[j] * inv;
}

// ---------------------------------------------------------------------------
extern "C" void kernel_launch(void* const* d_in, const int* in_sizes, int n_in,
                              void* d_out, int out_size, void* d_ws, size_t ws_size,
                              hipStream_t stream) {
    (void)in_sizes; (void)n_in; (void)out_size; (void)ws_size;
    const float* x    = (const float*)d_in[0];
    const int*   ei   = (const int*)d_in[1];
    const float* w_in = (const float*)d_in[2];
    const float* b_in = (const float*)d_in[3];
    const float* wl   = (const float*)d_in[4];
    const float* wr   = (const float*)d_in[5];
    const float* att  = (const float*)d_in[6];
    const float* bias = (const float*)d_in[7];
    const float* ln_g = (const float*)d_in[8];
    const float* ln_b = (const float*)d_in[9];
    float* out = (float*)d_out;

    char* base = (char*)d_ws; size_t off = 0;
    auto alloc = [&](size_t bytes) -> void* {
        void* r = base + off;
        off = (off + bytes + 255) & ~(size_t)255;
        return r;
    };
    unsigned short* xb    = (unsigned short*)alloc((size_t)NN * IND * 2);
    float*          h     = (float*)alloc((size_t)NN * DD * 4);
    unsigned short* hb    = (unsigned short*)alloc((size_t)NN * DD * 2);
    float*          xlbuf = (float*)alloc((size_t)NN * DD * 4);
    float*          xrbuf = (float*)alloc((size_t)NN * DD * 4);
    float*          ebuf  = (float*)alloc((size_t)E2 * HH * 4);
    int*            deg   = (int*)alloc((size_t)NN * 4);
    int*            rowoff= (int*)alloc((size_t)NN * 4);
    int*            cursor= (int*)alloc((size_t)NN * 4);
    int*            elist = (int*)alloc((size_t)E2 * 4);
    unsigned short* w_inT = (unsigned short*)alloc((size_t)IND * DD * 2);
    unsigned short* wlT   = (unsigned short*)alloc((size_t)LL * DD * DD * 2);
    unsigned short* wrT   = (unsigned short*)alloc((size_t)LL * DD * DD * 2);

    const int B = 256;

    // CSR by destination (topology constant across layers)
    zero_int2_k<<<(NN + B - 1) / B, B, 0, stream>>>(deg, cursor, NN);
    csr_count_k<<<(E2 + B - 1) / B, B, 0, stream>>>(ei, deg);
    csr_scan_k<<<1, B, 0, stream>>>(deg, rowoff);
    csr_fill_k<<<(E2 + B - 1) / B, B, 0, stream>>>(ei, rowoff, cursor, elist);

    // weights -> bf16 transposed
    wT_bf16_k<<<(IND * DD + B - 1) / B, B, 0, stream>>>(w_in, w_inT, IND, DD);
    for (int l = 0; l < LL; ++l) {
        wT_bf16_k<<<(DD * DD + B - 1) / B, B, 0, stream>>>(
            wl + (size_t)l * DD * DD, wlT + (size_t)l * DD * DD, DD, DD);
        wT_bf16_k<<<(DD * DD + B - 1) / B, B, 0, stream>>>(
            wr + (size_t)l * DD * DD, wrT + (size_t)l * DD * DD, DD, DD);
    }

    // h = x @ w_in + b_in
    f32_to_bf16_k<<<(NN * IND + B - 1) / B, B, 0, stream>>>(x, xb, NN * IND);
    dim3 ggrid(NN / 32, 1);   // 625 blocks, 8 waves, 2 M-tiles x 16 N-tiles
    gemm_bf16_wmma<<<ggrid, 256, 0, stream>>>(xb, w_inT, b_in, h, DD, IND);

    for (int l = 0; l < LL; ++l) {
        f32_to_bf16_k<<<(NN * DD + B - 1) / B, B, 0, stream>>>(h, hb, NN * DD);
        gemm_bf16_wmma<<<ggrid, 256, 0, stream>>>(
            hb, wlT + (size_t)l * DD * DD, nullptr, xlbuf, DD, DD);
        gemm_bf16_wmma<<<ggrid, 256, 0, stream>>>(
            hb, wrT + (size_t)l * DD * DD, nullptr, xrbuf, DD, DD);
        gat_node_k<<<(NN + 7) / 8, 256, 0, stream>>>(
            xlbuf, xrbuf, ei, rowoff, deg, elist,
            att + (size_t)l * HH * CC, bias + (size_t)l * DD,
            ln_g + (size_t)l * DD, ln_b + (size_t)l * DD, h, ebuf);
    }
    final_norm_k<<<(NN + 7) / 8, 256, 0, stream>>>(h, out);
}
